// EMTransformerEncoder_34385508172273
// MI455X (gfx1250) — matrix operand order repaired and checked by reading, not compiled, not run on hardware
//
#include <hip/hip_runtime.h>

#define B_ 2
#define N_ 8000
#define D_ 256
#define H_ 8
#define DH_ 32
#define F_ 1024
#define KMAX 3200
#define NLAYERS 4

typedef _Float16 v8h  __attribute__((ext_vector_type(8)));
typedef _Float16 v16h __attribute__((ext_vector_type(16)));
typedef float    v8f  __attribute__((ext_vector_type(8)));

union VH { v16h v; v8h h2[2]; };

__device__ __forceinline__ unsigned f2key(float f) {
  unsigned u = __float_as_uint(f);
  return (u & 0x80000000u) ? ~u : (u | 0x80000000u);
}

__device__ __forceinline__ float gelu_tanh(float x) {
  float x3 = x * x * x;
  return 0.5f * x * (1.0f + tanhf(0.7978845608028654f * (x + 0.044715f * x3)));
}

// ---------------- top-k: radix select threshold (1 block per batch) ----------
__global__ void topk_threshold_kernel(const float* __restrict__ sal, int k,
                                      unsigned* __restrict__ thr) {
  int b = blockIdx.x;
  const float* s = sal + (size_t)b * N_;
  __shared__ unsigned hist[256];
  __shared__ unsigned pfx_sh;
  __shared__ int rem_sh;
  unsigned prefix = 0;
  int rem = k;
  for (int pass = 3; pass >= 0; --pass) {
    for (int i = threadIdx.x; i < 256; i += blockDim.x) hist[i] = 0;
    __syncthreads();
    unsigned shift = (unsigned)pass * 8u;
    unsigned mask_hi = (pass == 3) ? 0u : (0xFFFFFFFFu << (shift + 8u));
    for (int i = threadIdx.x; i < N_; i += blockDim.x) {
      unsigned key = f2key(s[i]);
      if ((key & mask_hi) == prefix) atomicAdd(&hist[(key >> shift) & 255u], 1u);
    }
    __syncthreads();
    if (threadIdx.x == 0) {
      int r = rem;
      int bin = 255;
      for (; bin > 0; --bin) {
        int c = (int)hist[bin];
        if (r <= c) break;
        r -= c;
      }
      pfx_sh = prefix | ((unsigned)bin << shift);
      rem_sh = r;
    }
    __syncthreads();
    prefix = pfx_sh;
    rem = rem_sh;
    __syncthreads();
  }
  if (threadIdx.x == 0) thr[b] = prefix;
}

// ------------- deterministic compaction via block scan (1 block per batch) ---
__global__ void topk_compact_kernel(const float* __restrict__ sal,
                                    const unsigned* __restrict__ thr,
                                    int k, int* __restrict__ idx) {
  int b = blockIdx.x;
  unsigned T = thr[b];
  const float* s = sal + (size_t)b * N_;
  __shared__ int sdata[1024];
  int tid = threadIdx.x;
  // phase 1: g = count(key > T)
  int cnt = 0;
  for (int i = tid; i < N_; i += 1024) cnt += (f2key(s[i]) > T) ? 1 : 0;
  sdata[tid] = cnt;
  __syncthreads();
  for (int off = 512; off > 0; off >>= 1) {
    if (tid < off) sdata[tid] += sdata[tid + off];
    __syncthreads();
  }
  int g = sdata[0];
  __syncthreads();
  // phase 2: ordered chunked scan, assign slots
  int base_gt = 0, base_eq = 0;
  for (int c0 = 0; c0 < N_; c0 += 1024) {
    int i = c0 + tid;
    int pg = 0, pe = 0;
    if (i < N_) {
      unsigned key = f2key(s[i]);
      pg = (key > T) ? 1 : 0;
      pe = (key == T) ? 1 : 0;
    }
    int val = (pg << 16) | pe;
    sdata[tid] = val;
    __syncthreads();
    for (int off = 1; off < 1024; off <<= 1) {
      int t = (tid >= off) ? sdata[tid - off] : 0;
      __syncthreads();
      sdata[tid] += t;
      __syncthreads();
    }
    int incl = sdata[tid];
    int total = sdata[1023];
    __syncthreads();
    int excl = incl - val;
    if (pg) idx[b * KMAX + base_gt + (excl >> 16)] = i;
    if (pe) {
      int slot = g + base_eq + (excl & 0xFFFF);
      if (slot < k) idx[b * KMAX + slot] = i;
    }
    base_gt += (total >> 16);
    base_eq += (total & 0xFFFF);
  }
}

// ---------------- gather / scatter -------------------------------------------
__global__ void gather_kernel(const float* __restrict__ feat, const int* __restrict__ idx,
                              float* __restrict__ x, int kc) {
  int b = blockIdx.y;
  int t = blockIdx.x * blockDim.x + threadIdx.x;
  if (t >= kc * 64) return;
  int row = t >> 6, c = t & 63;
  int src = idx[b * KMAX + row];
  const float4* fp = (const float4*)(feat + (size_t)b * N_ * D_ + (size_t)src * D_);
  float4* xp = (float4*)(x + (size_t)b * KMAX * D_ + (size_t)row * D_);
  xp[c] = fp[c];
}

__global__ void scatter_kernel(const float* __restrict__ x, const int* __restrict__ idx,
                               float* __restrict__ feat, int kc) {
  int b = blockIdx.y;
  int t = blockIdx.x * blockDim.x + threadIdx.x;
  if (t >= kc * 64) return;
  int row = t >> 6, c = t & 63;
  int dst = idx[b * KMAX + row];
  const float4* xp = (const float4*)(x + (size_t)b * KMAX * D_ + (size_t)row * D_);
  float4* fp = (float4*)(feat + (size_t)b * N_ * D_ + (size_t)dst * D_);
  fp[c] = xp[c];
}

// ---------------- layernorm: one wave (32 lanes) per row ---------------------
__global__ void ln_kernel(const float* __restrict__ x, const float* __restrict__ g,
                          const float* __restrict__ bta, _Float16* __restrict__ h, int kc) {
  int wave = threadIdx.x >> 5, lane = threadIdx.x & 31;
  int row = blockIdx.x * 8 + wave;
  int b = blockIdx.y;
  if (row >= kc) return;
  const float* xr = x + (size_t)b * KMAX * D_ + (size_t)row * D_;
  float v[8];
  float sum = 0.f;
#pragma unroll
  for (int j = 0; j < 8; ++j) { v[j] = xr[lane + j * 32]; sum += v[j]; }
#pragma unroll
  for (int off = 16; off >= 1; off >>= 1) sum += __shfl_xor(sum, off, 32);
  float mu = sum * (1.0f / (float)D_);
  float var = 0.f;
#pragma unroll
  for (int j = 0; j < 8; ++j) { float d = v[j] - mu; var += d * d; }
#pragma unroll
  for (int off = 16; off >= 1; off >>= 1) var += __shfl_xor(var, off, 32);
  float rs = rsqrtf(var * (1.0f / (float)D_) + 1e-5f);
  _Float16* hr = h + (size_t)b * KMAX * D_ + (size_t)row * D_;
#pragma unroll
  for (int j = 0; j < 8; ++j) {
    int c = lane + j * 32;
    hr[c] = (_Float16)((v[j] - mu) * rs * g[c] + bta[c]);
  }
}

// --------- weight fp32 [L][K][N] -> f16 transposed [L][N][K] -----------------
__global__ void wconv_kernel(const float* __restrict__ W, _Float16* __restrict__ Wt,
                             int K, int Nn) {
  int l = blockIdx.y;
  const float* Ws = W + (size_t)l * K * Nn;
  _Float16* Wd = Wt + (size_t)l * K * Nn;
  int t = blockIdx.x * blockDim.x + threadIdx.x;
  if (t >= K * Nn) return;
  int n = t / K, kk = t % K;
  Wd[t] = (_Float16)Ws[(size_t)kk * Nn + n];
}

// ---------------- generic WMMA GEMM: C = act(A[M,K] @ Wt[N,K]^T) -------------
// mode 0: out f16 row-major [KMAX,Nd]; mode 1: out f16 transposed [Nd,KMAX];
// mode 2: fp32 residual add into x [KMAX,256]
__global__ void gemm16_kernel(const _Float16* __restrict__ A, const _Float16* __restrict__ Wt,
                              int M, int Kd, int Nd, int mode, int act,
                              _Float16* __restrict__ outH, float* __restrict__ outF) {
  int wave = threadIdx.x >> 5, lane = threadIdx.x & 31;
  int ntiles = Nd >> 4;
  int mtiles = (M + 15) >> 4;
  int tile = blockIdx.x * 8 + wave;
  if (tile >= ntiles * mtiles) return;
  int mt = tile / ntiles, nt = tile % ntiles;
  int b = blockIdx.y;
  int m0 = mt << 4, n0 = nt << 4;
  int col = lane & 15, half = lane >> 4;
  const _Float16* arow = A + (size_t)b * KMAX * Kd + (size_t)(m0 + col) * Kd + half * 8;
  const _Float16* brow = Wt + (size_t)(n0 + col) * Kd + half * 8;
  v8f acc = {};
  for (int kc = 0; kc < Kd; kc += 32) {
    VH a, w;
    a.h2[0] = *(const v8h*)(arow + kc);
    a.h2[1] = *(const v8h*)(arow + kc + 16);
    w.h2[0] = *(const v8h*)(brow + kc);
    w.h2[1] = *(const v8h*)(brow + kc + 16);
    __builtin_prefetch(brow + kc + 64, 0, 1);
    acc = __builtin_amdgcn_wmma_f32_16x16x32_f16(false, a.v, false, w.v, (short)0, acc,
                                                 false, false);
  }
#pragma unroll
  for (int r = 0; r < 8; ++r) {
    int mrow = m0 + r + half * 8;
    if (mrow >= M) continue;
    float vv = acc[r];
    if (act) vv = gelu_tanh(vv);
    int nc = n0 + col;
    if (mode == 0)
      outH[(size_t)b * KMAX * Nd + (size_t)mrow * Nd + nc] = (_Float16)vv;
    else if (mode == 1)
      outH[(size_t)b * Nd * KMAX + (size_t)nc * KMAX + mrow] = (_Float16)vv;
    else
      outF[(size_t)b * KMAX * D_ + (size_t)mrow * D_ + nc] += vv;
  }
}

// ---------------- flash-style attention: 1 wave = 16 queries x 1 head --------
__global__ void attn_kernel(const _Float16* __restrict__ qb, const _Float16* __restrict__ kb,
                            const _Float16* __restrict__ vtb, _Float16* __restrict__ ob,
                            int kc) {
  __shared__ _Float16 psh[8][16][32];  // wave-private P tiles
  int wave = threadIdx.x >> 5, lane = threadIdx.x & 31;
  int hh = blockIdx.y, b = blockIdx.z;
  int q0 = (blockIdx.x * 8 + wave) * 16;
  if (q0 >= kc) return;
  int col = lane & 15, half = lane >> 4;
  const _Float16* Qp = qb + (size_t)b * KMAX * D_;
  const _Float16* Kp = kb + (size_t)b * KMAX * D_;
  const _Float16* Vp = vtb + (size_t)b * D_ * KMAX + (size_t)hh * DH_ * KMAX;
  VH aq;
  {
    const _Float16* qr = Qp + (size_t)(q0 + col) * D_ + hh * DH_ + half * 8;
    aq.h2[0] = *(const v8h*)qr;
    aq.h2[1] = *(const v8h*)(qr + 16);
  }
  v8f o0 = {}, o1 = {};
  v8f zero = {};
  float mrow[8], lrow[8];
#pragma unroll
  for (int r = 0; r < 8; ++r) { mrow[r] = -1e30f; lrow[r] = 0.f; }
  const float scale = 0.17677669529663687f;  // 1/sqrt(32)
  int kpad = (kc + 31) & ~31;
  for (int kcb = 0; kcb < kpad; kcb += 32) {
    VH bk0, bk1;
    const _Float16* kr = Kp + (size_t)(kcb + col) * D_ + hh * DH_ + half * 8;
    bk0.h2[0] = *(const v8h*)kr;
    bk0.h2[1] = *(const v8h*)(kr + 16);
    const _Float16* kr2 = kr + 16 * D_;
    bk1.h2[0] = *(const v8h*)kr2;
    bk1.h2[1] = *(const v8h*)(kr2 + 16);
    v8f s0 = __builtin_amdgcn_wmma_f32_16x16x32_f16(false, aq.v, false, bk0.v, (short)0,
                                                    zero, false, false);
    v8f s1 = __builtin_amdgcn_wmma_f32_16x16x32_f16(false, aq.v, false, bk1.v, (short)0,
                                                    zero, false, false);
    float cm0 = (kcb + col < kc) ? 0.f : -1e30f;
    float cm1 = (kcb + 16 + col < kc) ? 0.f : -1e30f;
#pragma unroll
    for (int r = 0; r < 8; ++r) {
      float a0 = s0[r] * scale + cm0;
      float a1 = s1[r] * scale + cm1;
      float mx = fmaxf(a0, a1);
#pragma unroll
      for (int off = 8; off >= 1; off >>= 1) mx = fmaxf(mx, __shfl_xor(mx, off, 32));
      float mn = fmaxf(mrow[r], mx);
      float corr = expf(mrow[r] - mn);
      float p0 = expf(a0 - mn);
      float p1 = expf(a1 - mn);
      float rs = p0 + p1;
#pragma unroll
      for (int off = 8; off >= 1; off >>= 1) rs += __shfl_xor(rs, off, 32);
      lrow[r] = lrow[r] * corr + rs;
      mrow[r] = mn;
      o0[r] *= corr;
      o1[r] *= corr;
      psh[wave][r + half * 8][col] = (_Float16)p0;
      psh[wave][r + half * 8][16 + col] = (_Float16)p1;
    }
    // LDS ops are in-order within a wave; barrier stops compiler reordering.
    asm volatile("" ::: "memory");
    VH ap, av0, av1;
    ap.h2[0] = *(const v8h*)&psh[wave][col][half * 8];
    ap.h2[1] = *(const v8h*)&psh[wave][col][16 + half * 8];
    const _Float16* vr0 = Vp + (size_t)col * KMAX + kcb + half * 8;
    av0.h2[0] = *(const v8h*)vr0;
    av0.h2[1] = *(const v8h*)(vr0 + 16);
    const _Float16* vr1 = Vp + (size_t)(16 + col) * KMAX + kcb + half * 8;
    av1.h2[0] = *(const v8h*)vr1;
    av1.h2[1] = *(const v8h*)(vr1 + 16);
    o0 = __builtin_amdgcn_wmma_f32_16x16x32_f16(false, ap.v, false, av0.v, (short)0, o0,
                                                false, false);
    o1 = __builtin_amdgcn_wmma_f32_16x16x32_f16(false, ap.v, false, av1.v, (short)0, o1,
                                                false, false);
  }
#pragma unroll
  for (int r = 0; r < 8; ++r) {
    int qr = q0 + r + half * 8;
    if (qr >= kc) continue;
    float inv = 1.0f / lrow[r];
    _Float16* op = ob + (size_t)b * KMAX * D_ + (size_t)qr * D_ + hh * DH_;
    op[col] = (_Float16)(o0[r] * inv);
    op[16 + col] = (_Float16)(o1[r] * inv);
  }
}

// ---------------------------------------------------------------------------
extern "C" void kernel_launch(void* const* d_in, const int* in_sizes, int n_in,
                              void* d_out, int out_size, void* d_ws, size_t ws_size,
                              hipStream_t stream) {
  (void)in_sizes; (void)n_in; (void)out_size; (void)ws_size;
  const float* features = (const float*)d_in[0];
  const float* salience = (const float*)d_in[1];
  const float* Wq = (const float*)d_in[2];
  const float* Wk = (const float*)d_in[3];
  const float* Wv = (const float*)d_in[4];
  const float* Wo = (const float*)d_in[5];
  const float* W1 = (const float*)d_in[6];
  const float* W2 = (const float*)d_in[7];
  const float* ln1g = (const float*)d_in[8];
  const float* ln1b = (const float*)d_in[9];
  const float* ln2g = (const float*)d_in[10];
  const float* ln2b = (const float*)d_in[11];
  float* out = (float*)d_out;

  char* ws = (char*)d_ws;
  size_t off = 0;
  auto take = [&](size_t bytes) -> char* {
    char* p = ws + off;
    off = (off + bytes + 255) & ~(size_t)255;
    return p;
  };
  int* idx = (int*)take((size_t)B_ * KMAX * 4);
  unsigned* thr = (unsigned*)take((size_t)B_ * 4);
  float* x = (float*)take((size_t)B_ * KMAX * D_ * 4);
  _Float16* h = (_Float16*)take((size_t)B_ * KMAX * D_ * 2);
  _Float16* qb = (_Float16*)take((size_t)B_ * KMAX * D_ * 2);
  _Float16* kbf = (_Float16*)take((size_t)B_ * KMAX * D_ * 2);
  _Float16* vt = (_Float16*)take((size_t)B_ * D_ * KMAX * 2);
  _Float16* ob = (_Float16*)take((size_t)B_ * KMAX * D_ * 2);
  _Float16* ff = (_Float16*)take((size_t)B_ * KMAX * F_ * 2);
  _Float16* wqt = (_Float16*)take((size_t)NLAYERS * D_ * D_ * 2);
  _Float16* wkt = (_Float16*)take((size_t)NLAYERS * D_ * D_ * 2);
  _Float16* wvt = (_Float16*)take((size_t)NLAYERS * D_ * D_ * 2);
  _Float16* wot = (_Float16*)take((size_t)NLAYERS * D_ * D_ * 2);
  _Float16* w1t = (_Float16*)take((size_t)NLAYERS * D_ * F_ * 2);
  _Float16* w2t = (_Float16*)take((size_t)NLAYERS * F_ * D_ * 2);

  hipMemcpyAsync(out, features, (size_t)B_ * N_ * D_ * 4, hipMemcpyDeviceToDevice, stream);

  dim3 blk(256);
  wconv_kernel<<<dim3((D_ * D_ + 255) / 256, NLAYERS), blk, 0, stream>>>(Wq, wqt, D_, D_);
  wconv_kernel<<<dim3((D_ * D_ + 255) / 256, NLAYERS), blk, 0, stream>>>(Wk, wkt, D_, D_);
  wconv_kernel<<<dim3((D_ * D_ + 255) / 256, NLAYERS), blk, 0, stream>>>(Wv, wvt, D_, D_);
  wconv_kernel<<<dim3((D_ * D_ + 255) / 256, NLAYERS), blk, 0, stream>>>(Wo, wot, D_, D_);
  wconv_kernel<<<dim3((D_ * F_ + 255) / 256, NLAYERS), blk, 0, stream>>>(W1, w1t, D_, F_);
  wconv_kernel<<<dim3((F_ * D_ + 255) / 256, NLAYERS), blk, 0, stream>>>(W2, w2t, F_, D_);

  const int kls[NLAYERS] = {3200, 2400, 2000, 1600};
  for (int l = 0; l < NLAYERS; ++l) {
    int kc = kls[l];
    topk_threshold_kernel<<<B_, 1024, 0, stream>>>(salience, kc, thr);
    topk_compact_kernel<<<B_, 1024, 0, stream>>>(salience, thr, kc, idx);
    gather_kernel<<<dim3((kc * 64 + 255) / 256, B_), blk, 0, stream>>>(out, idx, x, kc);
    ln_kernel<<<dim3((kc + 7) / 8, B_), blk, 0, stream>>>(x, ln1g + l * D_, ln1b + l * D_, h, kc);
    int tilesD = ((kc + 15) / 16) * (D_ / 16);
    dim3 ggD((tilesD + 7) / 8, B_);
    gemm16_kernel<<<ggD, blk, 0, stream>>>(h, wqt + (size_t)l * D_ * D_, kc, D_, D_, 0, 0, qb, nullptr);
    gemm16_kernel<<<ggD, blk, 0, stream>>>(h, wkt + (size_t)l * D_ * D_, kc, D_, D_, 0, 0, kbf, nullptr);
    gemm16_kernel<<<ggD, blk, 0, stream>>>(h, wvt + (size_t)l * D_ * D_, kc, D_, D_, 1, 0, vt, nullptr);
    attn_kernel<<<dim3((kc + 127) / 128, H_, B_), blk, 0, stream>>>(qb, kbf, vt, ob, kc);
    gemm16_kernel<<<ggD, blk, 0, stream>>>(ob, wot + (size_t)l * D_ * D_, kc, D_, D_, 2, 0, nullptr, x);
    ln_kernel<<<dim3((kc + 7) / 8, B_), blk, 0, stream>>>(x, ln2g + l * D_, ln2b + l * D_, h, kc);
    int tilesF = ((kc + 15) / 16) * (F_ / 16);
    gemm16_kernel<<<dim3((tilesF + 7) / 8, B_), blk, 0, stream>>>(h, w1t + (size_t)l * D_ * F_, kc, D_, F_, 0, 1, ff, nullptr);
    gemm16_kernel<<<ggD, blk, 0, stream>>>(ff, w2t + (size_t)l * F_ * D_, kc, F_, D_, 2, 0, nullptr, x);
    scatter_kernel<<<dim3((kc * 64 + 255) / 256, B_), blk, 0, stream>>>(x, idx, out, kc);
  }
}
